// Transformer_Layer_63273458204891
// MI455X (gfx1250) — compile-verified, matrix operand order checked by reading, not run on hardware
//
#include <hip/hip_runtime.h>
#include <hip/hip_bf16.h>
#include <math.h>

// ---------------------------------------------------------------------------
// MI455X (gfx1250) implementation.
// Dominant cost: five 8192x2048x2048 GEMMs (~344 GFLOP) -> v_wmma_f32_16x16x32_f16
// with f32 accumulation, 128x128 block tiles, K=32 LDS staging, 8 waves/block
// (wave32). Weights pre-transposed to N-major f16 so A and B^T fragments are
// two contiguous 16B loads per lane, matching the CDNA5 16-bit A-matrix VGPR
// layout (lanes 0-15: K 0..7/16..23, lanes 16-31: K 8..15/24..31).
// ---------------------------------------------------------------------------

typedef __attribute__((ext_vector_type(16))) _Float16 v16h;
typedef __attribute__((ext_vector_type(8)))  _Float16 v8h;
typedef __attribute__((ext_vector_type(8)))  float    v8f;

#define B_     16
#define P_     16
#define PS_    16
#define NV_    32
#define F_     128
#define L_     256
#define D_     2048
#define NH_    8
#define DH_    256
#define BN_    512      // B*NV
#define MROWS_ 8192     // BN*P

// Load a 16x32 f16 WMMA A/B fragment from a row-major (ld halves) buffer.
// lane l: m = l&15, hi = l>>4.  VGPR packing per ISA 7.12.2:
//   halves 0..7  -> K = hi*8 + 0..7
//   halves 8..15 -> K = 16 + hi*8 + 0..7
__device__ __forceinline__ v16h load_frag(const _Float16* base, int ld, int lane) {
    int m  = lane & 15;
    int hi = (lane >> 4) & 1;
    const _Float16* r = base + m * ld + hi * 8;
    v8h lo = *(const v8h*)(r);
    v8h hh = *(const v8h*)(r + 16);
    v16h v;
#pragma unroll
    for (int i = 0; i < 8; ++i) { v[i] = lo[i]; v[8 + i] = hh[i]; }
    return v;
}

// ---------------------------------------------------------------------------
// Tiled GEMM: C[M,N] = A[M,K] * B[K,N] (+bias[N]) (+wpos[row%16]);
// A row-major f16, BT = B^T row-major f16 (N x K). Output f16 and/or f32.
// Block tile 128x128, K-step 32. 8 waves: wave = (msub<2 bits> | nsub).
// ---------------------------------------------------------------------------
__global__ __launch_bounds__(256) void gemm_bt_kernel(
    const _Float16* __restrict__ A, const _Float16* __restrict__ BT,
    const float* __restrict__ bias, const float* __restrict__ wpos,
    _Float16* __restrict__ outH, float* __restrict__ outF,
    int M, int N, int K)
{
    __shared__ __align__(16) _Float16 As [128 * 32];
    __shared__ __align__(16) _Float16 Bsh[128 * 32];

    int tid   = threadIdx.x;
    int lane  = tid & 31;
    int wave  = tid >> 5;
    int tileN = blockIdx.x * 128;
    int tileM = blockIdx.y * 128;
    int wm    = (wave & 3) * 32;   // 2 m-subtiles of 16
    int wn    = (wave >> 2) * 64;  // 4 n-subtiles of 16

    v8f acc[2][4];
#pragma unroll
    for (int i = 0; i < 2; ++i)
#pragma unroll
        for (int j = 0; j < 4; ++j) acc[i][j] = {};

    for (int k0 = 0; k0 < K; k0 += 32) {
        // cooperative tile load: 128 rows x 32 halves, 8 halves (16B) per chunk
#pragma unroll
        for (int pass = 0; pass < 2; ++pass) {
            int c   = tid + pass * 256;      // 0..511
            int row = c >> 2;
            int ko  = (c & 3) * 8;
            *(v8h*)&As [row * 32 + ko] = *(const v8h*)&A [(size_t)(tileM + row) * K + k0 + ko];
            *(v8h*)&Bsh[row * 32 + ko] = *(const v8h*)&BT[(size_t)(tileN + row) * K + k0 + ko];
        }
        if (k0 + 32 < K) {  // hint next K-tile into cache (global_prefetch_b8)
            __builtin_prefetch(&A [(size_t)(tileM + (tid >> 1)) * K + k0 + 32 + (tid & 1) * 16], 0, 1);
            __builtin_prefetch(&BT[(size_t)(tileN + (tid >> 1)) * K + k0 + 32 + (tid & 1) * 16], 0, 1);
        }
        __syncthreads();

        v16h a0 = load_frag(&As[(wm +  0) * 32], 32, lane);
        v16h a1 = load_frag(&As[(wm + 16) * 32], 32, lane);
#pragma unroll
        for (int j = 0; j < 4; ++j) {
            v16h b = load_frag(&Bsh[(wn + j * 16) * 32], 32, lane);
            acc[0][j] = __builtin_amdgcn_wmma_f32_16x16x32_f16(false, a0, false, b, (short)0, acc[0][j], false, false);
            acc[1][j] = __builtin_amdgcn_wmma_f32_16x16x32_f16(false, a1, false, b, (short)0, acc[1][j], false, false);
        }
        __syncthreads();
    }

    // epilogue: lane l holds rows (e + 8*(l>>4)), col (l&15) of each 16x16 tile
    int m16  = (lane >> 4) * 8;
    int col0 = lane & 15;
#pragma unroll
    for (int i = 0; i < 2; ++i) {
#pragma unroll
        for (int j = 0; j < 4; ++j) {
            int gcol = tileN + wn + j * 16 + col0;
            float bb = bias ? bias[gcol] : 0.0f;
#pragma unroll
            for (int e = 0; e < 8; ++e) {
                int grow = tileM + wm + i * 16 + m16 + e;
                float v = acc[i][j][e] + bb + (wpos ? wpos[grow & 15] : 0.0f);
                if (outH) outH[(size_t)grow * N + gcol] = (_Float16)v;
                if (outF) outF[(size_t)grow * N + gcol] = v;
            }
        }
    }
}

// ---------------------------------------------------------------------------
// fp32 (K=2048 x N=2048) -> f16 transposed (N-major) weight conversion
// ---------------------------------------------------------------------------
__global__ __launch_bounds__(256) void cvt_transpose_kernel(
    const float* __restrict__ W, _Float16* __restrict__ WT)
{
    __shared__ float tile[32][33];
    int tx = threadIdx.x & 31, ty = threadIdx.x >> 5;
    int n0 = blockIdx.x * 32, k0 = blockIdx.y * 32;
#pragma unroll
    for (int i = 0; i < 32; i += 8)
        tile[ty + i][tx] = W[(size_t)(k0 + ty + i) * 2048 + n0 + tx];
    __syncthreads();
#pragma unroll
    for (int i = 0; i < 32; i += 8)
        WT[(size_t)(n0 + ty + i) * 2048 + k0 + tx] = (_Float16)tile[tx][ty + i];
}

// xu[(bn*16+p)*2048 + f*16+ps] = x[b, p*16+ps, n, f]  (f32 -> f16)
__global__ __launch_bounds__(256) void cvt_xu_kernel(
    const float* __restrict__ x, _Float16* __restrict__ xu)
{
    int idx = blockIdx.x * 256 + threadIdx.x;
    int d = idx & 2047;
    int pr = idx >> 11;
    int p = pr & 15, bn = pr >> 4;
    int b = bn >> 5, n = bn & 31;
    int f = d >> 4, ps = d & 15;
    xu[idx] = (_Float16)x[(((size_t)b * 256 + p * 16 + ps) * 32 + n) * 128 + f];
}

// ---------------------------------------------------------------------------
// Tiny prep kernels (noise-level FLOPs)
// ---------------------------------------------------------------------------
__global__ __launch_bounds__(256) void prep_m_kernel(
    const float* __restrict__ mem,
    const float* __restrict__ g1W, const float* __restrict__ g1b,
    const float* __restrict__ g2W, const float* __restrict__ g2b,
    const float* __restrict__ g3W, const float* __restrict__ g3b,
    float* __restrict__ m_out)
{
    __shared__ float h1[32 * 64];
    __shared__ float h2[32 * 64];
    int tid = threadIdx.x;
    for (int i = tid; i < 2048; i += 256) {
        int n = i >> 6, j = i & 63;
        float s = g1b[j];
        for (int d = 0; d < 16; ++d) s += mem[n * 16 + d] * g1W[d * 64 + j];
        h1[i] = tanhf(s);
    }
    __syncthreads();
    for (int i = tid; i < 2048; i += 256) {
        int n = i >> 6, j = i & 63;
        float s = g2b[j];
        for (int d = 0; d < 64; ++d) s += h1[n * 64 + d] * g2W[d * 64 + j];
        h2[i] = tanhf(s);
    }
    __syncthreads();
    for (int i = tid; i < 3200; i += 256) {
        int n = i / 100, j = i % 100;
        float s = g3b[j];
        for (int d = 0; d < 64; ++d) s += h2[n * 64 + d] * g3W[d * 100 + j];
        m_out[i] = s;
    }
}

// bias_d[k][n][f] and wdT[k][n][g][f] = Wd[k][n][f][g] (B^T layout, f16)
__global__ __launch_bounds__(256) void prep_wd_kernel(
    const float* __restrict__ m, const float* __restrict__ Bd,
    const float* __restrict__ Pd, const float* __restrict__ Qd,
    float* __restrict__ bias_d, _Float16* __restrict__ wdT)
{
    int k = blockIdx.x >> 5, n = blockIdx.x & 31;
    int tid = threadIdx.x;
    __shared__ float tmp[128 * 10];
    if (tid < 128) {
        float s = 0.0f;
        for (int mi = 0; mi < 100; ++mi) s += m[n * 100 + mi] * Bd[(k * 100 + mi) * 128 + tid];
        bias_d[(k * 32 + n) * 128 + tid] = s;
    }
    for (int i = tid; i < 1280; i += 256) {
        int f = i / 10, qi = i % 10;
        float s = 0.0f;
        for (int mi = 0; mi < 10; ++mi) s += Pd[(k * 128 + f) * 10 + mi] * m[n * 100 + mi * 10 + qi];
        tmp[i] = s;
    }
    __syncthreads();
    for (int i = tid; i < 16384; i += 256) {
        int f = i >> 7, g = i & 127;
        float s = 0.0f;
        for (int qi = 0; qi < 10; ++qi) s += tmp[f * 10 + qi] * Qd[(k * 10 + qi) * 128 + g];
        wdT[(((size_t)k * 32 + n) * 128 + g) * 128 + f] = (_Float16)s;
    }
}

__global__ __launch_bounds__(256) void prep_qemb_kernel(
    const float* __restrict__ ie, const float* __restrict__ embW,
    const float* __restrict__ embB, float* __restrict__ qemb)
{
    int p = blockIdx.x;
    for (int i = threadIdx.x; i < 4096; i += 256) {
        int n = i >> 7, f = i & 127;
        float s = embB[p * 128 + f];
        for (int d = 0; d < 16; ++d) s += ie[(p * 32 + n) * 16 + d] * embW[(p * 16 + d) * 128 + f];
        qemb[(p * 32 + n) * 128 + f] = s;
    }
}

// ---------------------------------------------------------------------------
// Intra attention: per (b,p), 8 n per block. key_/val_ = t @ Wd[k][n] via WMMA
// (t padded 17->32 rows), softmax over 17, Ps projections, -> intra[b][p][n][f]
// ---------------------------------------------------------------------------
__global__ __launch_bounds__(256) void intra_kernel(
    const float* __restrict__ x, const float* __restrict__ qemb,
    const _Float16* __restrict__ wdT, const float* __restrict__ bias_d,
    const float* __restrict__ Ps, const float* __restrict__ Bs,
    float* __restrict__ intra)
{
    __shared__ __align__(16) _Float16 tpad[32 * 128];
    __shared__ __align__(16) _Float16 kv[2][32 * 128];
    __shared__ float sco[2][17];
    __shared__ float attw[2][17];
    __shared__ float obuf[128];
    __shared__ float hbuf[128];

    int tid = threadIdx.x, lane = tid & 31, wave = tid >> 5;
    int chunk = blockIdx.x & 3;
    int bp = blockIdx.x >> 2;
    int b = bp >> 4, p = bp & 15;

    for (int nl = 0; nl < 8; ++nl) {
        int n = chunk * 8 + nl;
        // t = [qemb row ; 16 x rows ; zero pad]
        for (int i = tid; i < 32 * 128; i += 256) {
            int row = i >> 7, f = i & 127;
            float v = 0.0f;
            if (row == 0)       v = qemb[(p * 32 + n) * 128 + f];
            else if (row <= 16) v = x[(((size_t)b * 256 + p * 16 + (row - 1)) * 32 + n) * 128 + f];
            tpad[i] = (_Float16)v;
        }
        __syncthreads();
        // 32 output tiles: kk(2) x mt(2) x nt(8), 4 per wave; K = 128 (4 steps)
#pragma unroll
        for (int q = 0; q < 4; ++q) {
            int t  = wave * 4 + q;
            int kk = t >> 4;
            int mt = (t >> 3) & 1;
            int nt = t & 7;
            v8f acc = {};
#pragma unroll
            for (int ks = 0; ks < 4; ++ks) {
                v16h a = load_frag(&tpad[(mt * 16) * 128 + ks * 32], 128, lane);
                v16h bfr = load_frag(&wdT[(((size_t)kk * 32 + n) * 128 + nt * 16) * 128 + ks * 32], 128, lane);
                acc = __builtin_amdgcn_wmma_f32_16x16x32_f16(false, a, false, bfr, (short)0, acc, false, false);
            }
            int m16 = (lane >> 4) * 8;
            int col = nt * 16 + (lane & 15);
            float bb = bias_d[(kk * 32 + n) * 128 + col];
#pragma unroll
            for (int e = 0; e < 8; ++e) {
                int row = mt * 16 + m16 + e;
                if (row < 17) kv[kk][row * 128 + col] = (_Float16)(acc[e] + bb);
            }
        }
        __syncthreads();
        // scores: 2 half-heads x 17 keys, dot over 64
        if (tid < 34) {
            int kk = tid / 17, tt = tid % 17;
            float s = 0.0f;
            for (int d = 0; d < 64; ++d)
                s += (float)tpad[kk * 64 + d] * (float)kv[0][tt * 128 + kk * 64 + d];
            sco[kk][tt] = s * 0.125f;  // 1/sqrt(64)
        }
        __syncthreads();
        if (tid < 2) {
            float mx = -1e30f;
            for (int t2 = 0; t2 < 17; ++t2) mx = fmaxf(mx, sco[tid][t2]);
            float den = 0.0f;
            for (int t2 = 0; t2 < 17; ++t2) { float e = __expf(sco[tid][t2] - mx); attw[tid][t2] = e; den += e; }
            float inv = 1.0f / den;
            for (int t2 = 0; t2 < 17; ++t2) attw[tid][t2] *= inv;
        }
        __syncthreads();
        if (tid < 128) {
            int kk = tid >> 6;
            float o = 0.0f;
            for (int t2 = 0; t2 < 17; ++t2) o += attw[kk][t2] * (float)kv[1][t2 * 128 + tid];
            obuf[tid] = o;
        }
        __syncthreads();
        if (tid < 128) {
            float s = Bs[tid];
            for (int f = 0; f < 128; ++f) s += obuf[f] * Ps[f * 128 + tid];
            hbuf[tid] = fmaxf(s, 0.0f);
        }
        __syncthreads();
        if (tid < 128) {
            float s = Bs[128 + tid];
            for (int f = 0; f < 128; ++f) s += hbuf[f] * Ps[(128 + f) * 128 + tid];
            intra[(((size_t)b * 16 + p) * 32 + n) * 128 + tid] = s;
        }
        __syncthreads();
    }
}

// ---------------------------------------------------------------------------
// MHA: one block per (bn, head). q/k/v 16x256 f16 resident in LDS.
// ---------------------------------------------------------------------------
__global__ __launch_bounds__(256) void mha_kernel(
    const _Float16* __restrict__ qh, const _Float16* __restrict__ kh,
    const _Float16* __restrict__ vh, _Float16* __restrict__ aoh)
{
    __shared__ __align__(16) _Float16 qs[16 * 256];
    __shared__ __align__(16) _Float16 ks2[16 * 256];
    __shared__ __align__(16) _Float16 vs[16 * 256];
    __shared__ float sc[16 * 17];

    int tid = threadIdx.x;
    int h = blockIdx.x & 7;
    int bn = blockIdx.x >> 3;
    size_t base = (size_t)bn * 16 * 2048 + h * 256;

    for (int i = tid; i < 16 * 256; i += 256) {
        int row = i >> 8, d = i & 255;
        size_t g = base + (size_t)row * 2048 + d;
        qs[i] = qh[g]; ks2[i] = kh[g]; vs[i] = vh[g];
    }
    __syncthreads();
    {
        int qi = tid >> 4, kj = tid & 15;
        float s = 0.0f;
        for (int d = 0; d < 256; ++d) s += (float)qs[qi * 256 + d] * (float)ks2[kj * 256 + d];
        sc[qi * 17 + kj] = s * 0.0625f;  // 1/sqrt(256)
    }
    __syncthreads();
    {
        int qi = tid >> 4, kj = tid & 15;
        float mx = -1e30f;
        for (int k2 = 0; k2 < 16; ++k2) mx = fmaxf(mx, sc[qi * 17 + k2]);
        float den = 0.0f;
        for (int k2 = 0; k2 < 16; ++k2) den += __expf(sc[qi * 17 + k2] - mx);
        float pv = __expf(sc[qi * 17 + kj] - mx) / den;
        __syncthreads();
        sc[qi * 17 + kj] = pv;
    }
    __syncthreads();
    {
        int d = tid;
        for (int qi = 0; qi < 16; ++qi) {
            float a = 0.0f;
            for (int k2 = 0; k2 < 16; ++k2) a += sc[qi * 17 + k2] * (float)vs[k2 * 256 + d];
            aoh[base + (size_t)qi * 2048 + d] = (_Float16)a;
        }
    }
}

// ---------------------------------------------------------------------------
// Final: out = x + intra_out + inter ; result = out + gelu-FF(out)
// One block per (b,l,n) row of F=128.
// ---------------------------------------------------------------------------
__global__ __launch_bounds__(128) void final_kernel(
    const float* __restrict__ x, const float* __restrict__ intra,
    const float* __restrict__ linW, const float* __restrict__ linb,
    const float* __restrict__ interF,
    const float* __restrict__ f1W, const float* __restrict__ f1b,
    const float* __restrict__ f2W, const float* __restrict__ f2b,
    float* __restrict__ out)
{
    __shared__ float row[128];
    __shared__ float h1[64];
    int f = threadIdx.x;
    int r = blockIdx.x;
    int n = r & 31, l = (r >> 5) & 255, b = r >> 13;

    float xv = x[(size_t)r * 128 + f];
    float io = linb[l];
    for (int p = 0; p < 16; ++p)
        io += intra[(((size_t)b * 16 + p) * 32 + n) * 128 + f] * linW[p * 256 + l];

    // inter = ao.reshape(b,nv,p,ps,f).reshape(b,ps*p,nv,f) inverse index map
    int g = l * 32 + n;
    int ps = g & 15; int q2 = g >> 4; int psrc = q2 & 15; int nsrc = q2 >> 4;
    float iv = interF[((size_t)(b * 32 + nsrc) * 16 + psrc) * 2048 + ps * 128 + f];

    float o = xv + io + iv;
    row[f] = o;
    __syncthreads();
    if (f < 64) {
        float s = f1b[f];
        for (int f2 = 0; f2 < 128; ++f2) s += row[f2] * f1W[f2 * 64 + f];
        h1[f] = 0.5f * s * (1.0f + erff(s * 0.70710678118f));  // exact gelu
    }
    __syncthreads();
    float ff = f2b[f];
    for (int j = 0; j < 64; ++j) ff += h1[j] * f2W[j * 128 + f];
    out[(size_t)r * 128 + f] = o + ff;
}

// ---------------------------------------------------------------------------
extern "C" void kernel_launch(void* const* d_in, const int* in_sizes, int n_in,
                              void* d_out, int out_size, void* d_ws, size_t ws_size,
                              hipStream_t stream) {
    (void)in_sizes; (void)n_in; (void)out_size; (void)ws_size;
    const float* x    = (const float*)d_in[0];
    const float* ie   = (const float*)d_in[1];
    const float* embW = (const float*)d_in[2];
    const float* embB = (const float*)d_in[3];
    const float* mem  = (const float*)d_in[4];
    const float* g1W  = (const float*)d_in[5];
    const float* g1b  = (const float*)d_in[6];
    const float* g2W  = (const float*)d_in[7];
    const float* g2b  = (const float*)d_in[8];
    const float* g3W  = (const float*)d_in[9];
    const float* g3b  = (const float*)d_in[10];
    const float* Pd   = (const float*)d_in[11];
    const float* Qd   = (const float*)d_in[12];
    const float* Bd   = (const float*)d_in[13];
    const float* Ps   = (const float*)d_in[14];
    const float* Bs   = (const float*)d_in[15];
    const float* linW = (const float*)d_in[16];
    const float* linb = (const float*)d_in[17];
    const float* embLW= (const float*)d_in[18];
    const float* embLb= (const float*)d_in[19];
    const float* Wpos = (const float*)d_in[20];
    const float* Wq   = (const float*)d_in[21];
    const float* bq   = (const float*)d_in[22];
    const float* Wk   = (const float*)d_in[23];
    const float* bk   = (const float*)d_in[24];
    const float* Wv   = (const float*)d_in[25];
    const float* bv   = (const float*)d_in[26];
    const float* Wo   = (const float*)d_in[27];
    const float* bo   = (const float*)d_in[28];
    const float* f1W  = (const float*)d_in[29];
    const float* f1b  = (const float*)d_in[30];
    const float* f2W  = (const float*)d_in[31];
    const float* f2b  = (const float*)d_in[32];
    float* outp = (float*)d_out;

    char* wsp = (char*)d_ws;
    auto alloc = [&](size_t bytes) -> void* {
        void* r = (void*)wsp;
        wsp += (bytes + 255) & ~(size_t)255;
        return r;
    };
    const size_t DD = (size_t)D_ * D_;
    _Float16* wT_embL = (_Float16*)alloc(DD * 2);
    _Float16* wT_q    = (_Float16*)alloc(DD * 2);
    _Float16* wT_k    = (_Float16*)alloc(DD * 2);
    _Float16* wT_v    = (_Float16*)alloc(DD * 2);
    _Float16* wT_o    = (_Float16*)alloc(DD * 2);
    _Float16* xu_h    = (_Float16*)alloc((size_t)MROWS_ * D_ * 2);
    _Float16* xe_h    = (_Float16*)alloc((size_t)MROWS_ * D_ * 2);
    _Float16* qh      = (_Float16*)alloc((size_t)MROWS_ * D_ * 2);
    _Float16* kh      = (_Float16*)alloc((size_t)MROWS_ * D_ * 2);
    _Float16* vh      = (_Float16*)alloc((size_t)MROWS_ * D_ * 2);
    _Float16* ao_h    = (_Float16*)alloc((size_t)MROWS_ * D_ * 2);
    float*    interF  = (float*)   alloc((size_t)MROWS_ * D_ * 4);
    float*    m_buf   = (float*)   alloc(32 * 100 * 4);
    float*    bias_d  = (float*)   alloc(2 * 32 * 128 * 4);
    _Float16* wdT     = (_Float16*)alloc((size_t)2 * 32 * 128 * 128 * 2);
    float*    qemb    = (float*)   alloc((size_t)16 * 32 * 128 * 4);
    float*    intra   = (float*)   alloc((size_t)16 * 16 * 32 * 128 * 4);

    dim3 tb(256);
    dim3 tg(D_ / 32, D_ / 32);
    cvt_transpose_kernel<<<tg, tb, 0, stream>>>(embLW, wT_embL);
    cvt_transpose_kernel<<<tg, tb, 0, stream>>>(Wq, wT_q);
    cvt_transpose_kernel<<<tg, tb, 0, stream>>>(Wk, wT_k);
    cvt_transpose_kernel<<<tg, tb, 0, stream>>>(Wv, wT_v);
    cvt_transpose_kernel<<<tg, tb, 0, stream>>>(Wo, wT_o);
    cvt_xu_kernel<<<(MROWS_ * D_) / 256, 256, 0, stream>>>(x, xu_h);

    prep_m_kernel<<<1, 256, 0, stream>>>(mem, g1W, g1b, g2W, g2b, g3W, g3b, m_buf);
    prep_wd_kernel<<<64, 256, 0, stream>>>(m_buf, Bd, Pd, Qd, bias_d, wdT);
    prep_qemb_kernel<<<16, 256, 0, stream>>>(ie, embW, embB, qemb);

    dim3 gg(D_ / 128, MROWS_ / 128);  // (16, 64)
    gemm_bt_kernel<<<gg, 256, 0, stream>>>(xu_h, wT_embL, embLb, Wpos, xe_h, nullptr, MROWS_, D_, D_);
    gemm_bt_kernel<<<gg, 256, 0, stream>>>(xe_h, wT_q, bq, nullptr, qh, nullptr, MROWS_, D_, D_);
    gemm_bt_kernel<<<gg, 256, 0, stream>>>(xe_h, wT_k, bk, nullptr, kh, nullptr, MROWS_, D_, D_);
    gemm_bt_kernel<<<gg, 256, 0, stream>>>(xe_h, wT_v, bv, nullptr, vh, nullptr, MROWS_, D_, D_);

    mha_kernel<<<BN_ * NH_, 256, 0, stream>>>(qh, kh, vh, ao_h);

    gemm_bt_kernel<<<gg, 256, 0, stream>>>(ao_h, wT_o, bo, nullptr, nullptr, interF, MROWS_, D_, D_);

    intra_kernel<<<16 * 16 * 4, 256, 0, stream>>>(x, qemb, wdT, bias_d, Ps, Bs, intra);

    final_kernel<<<B_ * L_ * NV_, 128, 0, stream>>>(x, intra, linW, linb, interF,
                                                    f1W, f1b, f2W, f2b, outp);
}
